// MultiHeadAttention_37658273251976
// MI455X (gfx1250) — matrix operand order repaired and checked by reading, not compiled, not run on hardware
//
#include <hip/hip_runtime.h>

// MultiHeadAttention for MI455X (gfx1250): bf16 WMMA, f32 accumulate,
// async-to-LDS K/V staging, permlane16 softmax reductions, exp2-domain
// softmax, mask-free fast path on interior key blocks.
// B=2, S=2048, D=512, H=8, DK=64.  Assumes ws_size >= ~31 MB.

#define B_   2
#define S_   2048
#define D_   512
#define H_   8
#define DK_  64

typedef __attribute__((ext_vector_type(16))) __bf16        v16bf;
typedef __attribute__((ext_vector_type(8)))  float         v8f;
typedef __attribute__((ext_vector_type(4)))  unsigned int  u32x4;
typedef __attribute__((ext_vector_type(2)))  unsigned int  u32x2;
typedef __attribute__((ext_vector_type(4)))  float         f32x4;

union Frag {
    v16bf        bf;
    u32x4        q[2];   // two 16B pieces
    unsigned int u[8];
};

// float -> bf16 bits, round-to-nearest-even (all values here are finite)
__device__ __forceinline__ unsigned short f2bf(float f) {
    unsigned int u = __builtin_bit_cast(unsigned int, f);
    u += 0x7FFFu + ((u >> 16) & 1u);
    return (unsigned short)(u >> 16);
}

__device__ __forceinline__ unsigned int pack2(float lo, float hi) {
    return (unsigned int)f2bf(lo) | ((unsigned int)f2bf(hi) << 16);
}

__device__ __forceinline__ v8f wmma_bf16(const Frag& a, const Frag& b, v8f c) {
    // D = A(16x32 bf16) * B(32x16 bf16) + C(16x16 f32)
    return __builtin_amdgcn_wmma_f32_16x16x32_bf16(
        false, a.bf, false, b.bf, (short)0, c, false, false);
}

// ---- VALU lane permutes: xor across the 16-lane row (halves of wave32) ----
__device__ __forceinline__ float plane16(float x, unsigned lo, unsigned hi) {
    int xi = __builtin_bit_cast(int, x);
    int r  = __builtin_amdgcn_permlane16(xi, xi, lo, hi, false, false);
    return __builtin_bit_cast(float, r);
}
__device__ __forceinline__ float rmax16(float x) {
    x = fmaxf(x, plane16(x, 0x67452301u, 0xEFCDAB89u)); // lane ^ 1
    x = fmaxf(x, plane16(x, 0x54761032u, 0xDCFE98BAu)); // lane ^ 2
    x = fmaxf(x, plane16(x, 0x32107654u, 0xBA98FEDCu)); // lane ^ 4
    x = fmaxf(x, plane16(x, 0xFEDCBA98u, 0x76543210u)); // lane ^ 8
    return x;
}
__device__ __forceinline__ float rsum16(float x) {
    x += plane16(x, 0x67452301u, 0xEFCDAB89u);
    x += plane16(x, 0x54761032u, 0xDCFE98BAu);
    x += plane16(x, 0x32107654u, 0xBA98FEDCu);
    x += plane16(x, 0xFEDCBA98u, 0x76543210u);
    return x;
}

// ---- async Global -> LDS (16B per lane), tracked by ASYNCcnt ----
__device__ __forceinline__ void async_ld16(unsigned lds_off, const void* g) {
    asm volatile("global_load_async_to_lds_b128 %0, %1, off"
                 :: "v"(lds_off), "v"(g) : "memory");
}
__device__ __forceinline__ void wait_async0() {
    asm volatile("s_wait_asynccnt 0x0" ::: "memory");
}

// ---------------------------------------------------------------- lens ----
__global__ void lens_kernel(const int* __restrict__ row_mask,
                            const int* __restrict__ col_mask,
                            int* __restrict__ lens) {
    const int* src = (blockIdx.x < 2 ? row_mask : col_mask) + (blockIdx.x & 1) * S_;
    __shared__ int red[256];
    int t = threadIdx.x, s = 0;
    for (int i = t; i < S_; i += 256) s += src[i];
    red[t] = s;
    __syncthreads();
    for (int off = 128; off > 0; off >>= 1) {
        if (t < off) red[t] += red[t + off];
        __syncthreads();
    }
    if (t == 0) lens[blockIdx.x] = red[0];
}

// ----------------------------------------------------------------- cvt ----
// f32 -> bf16, 4 elements per thread (all our sizes are multiples of 4)
__global__ void cvt4_kernel(const float* __restrict__ src,
                            unsigned short* __restrict__ dst, int n4) {
    int i = blockIdx.x * blockDim.x + threadIdx.x;
    if (i < n4) {
        f32x4 v = ((const f32x4*)src)[i];
        u32x2 o;
        o.x = pack2(v.x, v.y);
        o.y = pack2(v.z, v.w);
        ((u32x2*)dst)[i] = o;
    }
}

// ---------------------------------------------------------------- proj ----
// out[m][n] = sum_k A[m][k] * W[n][k] + bias[n]   (A @ W.T + b), A/W bf16
// OMODE: 0 = bf16 row-major (scaled), 2 = bf16 Vt[b][h][dk][s], 3 = f32 out
template <int OMODE>
__global__ void proj_kernel(const unsigned short* __restrict__ Abf16,
                            const unsigned short* __restrict__ Wbf,
                            const float* __restrict__ bias,
                            void* __restrict__ Out, float scale) {
    const int wave = (blockIdx.x * blockDim.x + threadIdx.x) >> 5;
    const int lane = threadIdx.x & 31;
    const int col  = lane & 15;
    const int half = lane >> 4;
    const int nt = wave & 31, mt = wave >> 5;     // 256 x 32 tiles
    const int m0 = mt * 16, n0 = nt * 16;
    const int mrow = m0 + col;
    const int ncol = n0 + col;

    auto load_ab = [&](int ks, Frag& a, Frag& b) {
        a.q[0] = *(const u32x4*)(Abf16 + mrow * D_ + ks + 8 * half);
        a.q[1] = *(const u32x4*)(Abf16 + mrow * D_ + ks + 16 + 8 * half);
        b.q[0] = *(const u32x4*)(Wbf + ncol * D_ + ks + 8 * half);
        b.q[1] = *(const u32x4*)(Wbf + ncol * D_ + ks + 16 + 8 * half);
    };

    v8f c = {};
    Frag a, b;
    load_ab(0, a, b);
#pragma unroll 4
    for (int ks = 0; ks < D_; ks += 32) {
        Frag an, bn;
        if (ks + 32 < D_) load_ab(ks + 32, an, bn);   // prefetch next step
        c = wmma_bf16(a, b, c);
        a = an; b = bn;
    }

    const float bv = bias[ncol];
#pragma unroll
    for (int j = 0; j < 8; ++j) {
        const int row = m0 + j + 8 * half;
        const float val = (c[j] + bv) * scale;
        if (OMODE == 0) {
            ((unsigned short*)Out)[row * D_ + ncol] = f2bf(val);
        } else if (OMODE == 2) {
            const int b_ = row >> 11, s_ = row & (S_ - 1);
            const int h_ = ncol >> 6, dk = ncol & 63;
            ((unsigned short*)Out)[(((b_ * H_ + h_) * DK_ + dk) << 11) + s_] = f2bf(val);
        } else {
            ((float*)Out)[row * D_ + ncol] = val;
        }
    }
}

// ---------------------------------------------------------------- attn ----
// Block = 4 waves = 4 consecutive 16-query tiles of ONE (b, h): waves share
// async-staged, double-buffered K/V tiles in LDS. Flash online softmax in
// exp2 domain (log2e folded into Q scale).
#define WPB 4
__global__ void __launch_bounds__(128)
attn_kernel(const unsigned short* __restrict__ Qbf,
            const unsigned short* __restrict__ Kbf,
            const unsigned short* __restrict__ Vt,
            const int* __restrict__ lens,
            unsigned short* __restrict__ Abf) {
    __shared__ alignas(16) unsigned short Ktile[2][32][72]; // 32 keys x 64 dk (+pad)
    __shared__ alignas(16) unsigned short Vtile[2][64][40]; // 64 dk x 32 keys (+pad)
    __shared__ alignas(16) unsigned short ptile[WPB][16][32];

    const int wv   = threadIdx.x >> 5;
    const int lane = threadIdx.x & 31;
    const int col  = lane & 15;
    const int half = lane >> 4;

    const int tile0 = blockIdx.x * WPB;          // 2048 tiles, 4 per block
    const int bh = tile0 >> 7;                   // uniform in block (128 % 4 == 0)
    const int h  = bh & 7, b = bh >> 3;
    const int qt = (tile0 & 127) + wv;
    const int q0 = qt * 16;
    const int row_len = lens[b];
    const int col_len = lens[2 + b];

    // block-uniform loop bound (causal bound of LAST tile in block + col pad)
    const int kend_blk = min((tile0 & 127) * 16 + WPB * 16, col_len);
    const int nb = (kend_blk + 31) >> 5;
    const int kend_w = min(q0 + 16, col_len);    // this wave's useful bound

    // Q A-fragments for dk = 0..31 / 32..63 (Q pre-scaled by log2e/8)
    const unsigned short* qrow = Qbf + ((size_t)(b * S_ + q0 + col)) * D_ + h * DK_;
    Frag aq0, aq1;
    aq0.q[0] = *(const u32x4*)(qrow + 16 * half);
    aq0.q[1] = *(const u32x4*)(qrow + 16 * half + 8);
    aq1.q[0] = *(const u32x4*)(qrow + 32 + 16 * half);
    aq1.q[1] = *(const u32x4*)(qrow + 32 + 16 * half + 8);

    const unsigned short* kgbase = Kbf + ((size_t)(b * S_)) * D_ + h * DK_;
    const unsigned short* vgbase = Vt + ((size_t)(bh * DK_) << 11);

    // cooperative async stage of one 32-key block (4 x b128 per wave)
    auto stage = [&](int kb, int buf) {
#pragma unroll
        for (int i = 0; i < 2; ++i) {            // K: 32 rows x 128B
            const int u = (wv * 2 + i) * 32 + lane;
            const int key = u >> 3, seg = (u & 7) * 8;
            async_ld16((unsigned)(size_t)&Ktile[buf][key][seg],
                       kgbase + (size_t)(kb + key) * D_ + seg);
        }
#pragma unroll
        for (int i = 0; i < 2; ++i) {            // V: 64 rows x 64B
            const int u = (wv * 2 + i) * 32 + lane;
            const int dk = u >> 2, seg = (u & 3) * 8;
            async_ld16((unsigned)(size_t)&Vtile[buf][dk][seg],
                       vgbase + ((size_t)dk << 11) + kb + seg);
        }
    };

    v8f o[4] = {v8f{}, v8f{}, v8f{}, v8f{}};
    float M[8], L[8];
#pragma unroll
    for (int j = 0; j < 8; ++j) { M[j] = -1e30f; L[j] = 0.f; }

    if (nb > 0) stage(0, 0);

    for (int ib = 0; ib < nb; ++ib) {
        const int kb = ib * 32;
        const int buf = ib & 1;
        wait_async0();          // my stage(ib) done
        __syncthreads();        // everyone's stage(ib) done; prev compute done

        if (kb < kend_w) {      // waves past their causal bound skip compute
            // ---- scores: S(16x32) = Q(16x64) @ K^T, two 16-key tiles ----
            Frag bk00, bk01, bk10, bk11;
            bk00.q[0] = *(const u32x4*)&Ktile[buf][col][16 * half];
            bk00.q[1] = *(const u32x4*)&Ktile[buf][col][16 * half + 8];
            bk01.q[0] = *(const u32x4*)&Ktile[buf][col][32 + 16 * half];
            bk01.q[1] = *(const u32x4*)&Ktile[buf][col][32 + 16 * half + 8];
            bk10.q[0] = *(const u32x4*)&Ktile[buf][16 + col][16 * half];
            bk10.q[1] = *(const u32x4*)&Ktile[buf][16 + col][16 * half + 8];
            bk11.q[0] = *(const u32x4*)&Ktile[buf][16 + col][32 + 16 * half];
            bk11.q[1] = *(const u32x4*)&Ktile[buf][16 + col][32 + 16 * half + 8];
            v8f s0 = {}, s1 = {};
            s0 = wmma_bf16(aq0, bk00, s0);
            s0 = wmma_bf16(aq1, bk01, s0);
            s1 = wmma_bf16(aq0, bk10, s1);
            s1 = wmma_bf16(aq1, bk11, s1);

            // ---- online softmax; interior blocks skip all masking ----
            float p0[8], p1[8];
            const bool full = (kb + 31 <= q0) && (kb + 31 < col_len) &&
                              (q0 + 15 < row_len);
            if (full) {
#pragma unroll
                for (int j = 0; j < 8; ++j) {
                    const float x0 = s0[j], x1 = s1[j];
                    const float bm = rmax16(fmaxf(x0, x1));
                    const float newM = fmaxf(M[j], bm);
                    const float alpha = exp2f(M[j] - newM);
                    const float e0 = exp2f(x0 - newM);
                    const float e1 = exp2f(x1 - newM);
                    L[j] = L[j] * alpha + rsum16(e0 + e1);
                    M[j] = newM;
                    o[0][j] *= alpha; o[1][j] *= alpha;
                    o[2][j] *= alpha; o[3][j] *= alpha;
                    p0[j] = e0; p1[j] = e1;
                }
            } else {
                const int key0 = kb + col, key1 = kb + 16 + col;
#pragma unroll
                for (int j = 0; j < 8; ++j) {
                    const int q = q0 + j + 8 * half;
                    const bool v0 = (key0 <= q) && (q < row_len) && (key0 < col_len);
                    const bool v1 = (key1 <= q) && (q < row_len) && (key1 < col_len);
                    const float x0 = v0 ? s0[j] : -1e30f;
                    const float x1 = v1 ? s1[j] : -1e30f;
                    const float bm = rmax16(fmaxf(x0, x1));
                    const float newM = fmaxf(M[j], bm);
                    const float alpha = exp2f(M[j] - newM);
                    const float e0 = v0 ? exp2f(x0 - newM) : 0.f;
                    const float e1 = v1 ? exp2f(x1 - newM) : 0.f;
                    L[j] = L[j] * alpha + rsum16(e0 + e1);
                    M[j] = newM;
                    o[0][j] *= alpha; o[1][j] *= alpha;
                    o[2][j] *= alpha; o[3][j] *= alpha;
                    p0[j] = e0; p1[j] = e1;
                }
            }

            // ---- transpose P (C layout -> A layout) via per-wave LDS ----
#pragma unroll
            for (int j = 0; j < 8; ++j) {
                const int r = j + 8 * half;
                ptile[wv][r][col]      = f2bf(p0[j]);
                ptile[wv][r][16 + col] = f2bf(p1[j]);
            }
            asm volatile("s_wait_dscnt 0" ::: "memory");
            Frag ap;
            ap.q[0] = *(const u32x4*)(&ptile[wv][col][8 * half]);
            ap.q[1] = *(const u32x4*)(&ptile[wv][col][16 + 8 * half]);

            // ---- o += P(16x32) @ V(32x64) from staged Vtile ----
#pragma unroll
            for (int t = 0; t < 4; ++t) {
                Frag bv;
                bv.q[0] = *(const u32x4*)&Vtile[buf][t * 16 + col][8 * half];
                bv.q[1] = *(const u32x4*)&Vtile[buf][t * 16 + col][16 + 8 * half];
                o[t] = wmma_bf16(ap, bv, o[t]);
            }
        }

        if (ib + 1 < nb) stage(ib + 1, buf ^ 1);
    }

    // ---- finalize: /rowsum, fully-masked rows -> 0 (== nan_to_num) ----
#pragma unroll
    for (int j = 0; j < 8; ++j) {
        const float inv = L[j] > 0.f ? 1.0f / L[j] : 0.f;
        const int row = q0 + j + 8 * half;
        unsigned short* orow = Abf + ((size_t)(b * S_ + row)) * D_ + h * DK_;
        orow[col]      = f2bf(o[0][j] * inv);
        orow[16 + col] = f2bf(o[1][j] * inv);
        orow[32 + col] = f2bf(o[2][j] * inv);
        orow[48 + col] = f2bf(o[3][j] * inv);
    }
}

// -------------------------------------------------------------- launch ----
extern "C" void kernel_launch(void* const* d_in, const int* in_sizes, int n_in,
                              void* d_out, int out_size, void* d_ws, size_t ws_size,
                              hipStream_t stream) {
    const float* in_Q  = (const float*)d_in[0];
    const float* in_K  = (const float*)d_in[1];
    const float* in_V  = (const float*)d_in[2];
    const int*   rpm   = (const int*)d_in[3];
    const int*   cpm   = (const int*)d_in[4];
    const float* W_Q   = (const float*)d_in[5];
    const float* b_Q   = (const float*)d_in[6];
    const float* W_K   = (const float*)d_in[7];
    const float* b_K   = (const float*)d_in[8];
    const float* W_V   = (const float*)d_in[9];
    const float* b_V   = (const float*)d_in[10];
    const float* W_O   = (const float*)d_in[11];
    const float* b_O   = (const float*)d_in[12];

    char* ws = (char*)d_ws;
    const size_t MB = 1024 * 1024;
    unsigned short* Qbf = (unsigned short*)(ws + 0 * MB);    // 4 MB
    unsigned short* Kbf = (unsigned short*)(ws + 4 * MB);    // 4 MB
    unsigned short* Vt  = (unsigned short*)(ws + 8 * MB);    // 4 MB  [b][h][dk][s]
    unsigned short* Abf = (unsigned short*)(ws + 12 * MB);   // 4 MB
    unsigned short* Xq  = (unsigned short*)(ws + 16 * MB);   // 4 MB bf16 activations
    unsigned short* Xk  = (unsigned short*)(ws + 20 * MB);
    unsigned short* Xv  = (unsigned short*)(ws + 24 * MB);
    unsigned short* Wq  = (unsigned short*)(ws + 28 * MB);   // 512 KB each
    unsigned short* Wk  = (unsigned short*)(ws + 28 * MB + 512 * 1024);
    unsigned short* Wv  = (unsigned short*)(ws + 29 * MB);
    unsigned short* Wo  = (unsigned short*)(ws + 29 * MB + 512 * 1024);
    int*            lens = (int*)(ws + 30 * MB);

    lens_kernel<<<4, 256, 0, stream>>>(rpm, cpm, lens);

    // one-time f32 -> bf16 conversions (weights + activations)
    const int WN4 = D_ * D_ / 4;                 // 65536
    const int XN4 = B_ * S_ * D_ / 4;            // 524288
    cvt4_kernel<<<(WN4 + 255) / 256, 256, 0, stream>>>(W_Q, Wq, WN4);
    cvt4_kernel<<<(WN4 + 255) / 256, 256, 0, stream>>>(W_K, Wk, WN4);
    cvt4_kernel<<<(WN4 + 255) / 256, 256, 0, stream>>>(W_V, Wv, WN4);
    cvt4_kernel<<<(WN4 + 255) / 256, 256, 0, stream>>>(W_O, Wo, WN4);
    cvt4_kernel<<<(XN4 + 255) / 256, 256, 0, stream>>>(in_Q, Xq, XN4);
    cvt4_kernel<<<(XN4 + 255) / 256, 256, 0, stream>>>(in_K, Xk, XN4);
    cvt4_kernel<<<(XN4 + 255) / 256, 256, 0, stream>>>(in_V, Xv, XN4);

    // projections: 8192 tiles / 8 waves per 256-thread block = 1024 blocks
    const float scale_q = 0.125f * 1.44269504088896340736f; // log2e/sqrt(64)
    proj_kernel<0><<<1024, 256, 0, stream>>>(Xq, Wq, b_Q, Qbf, scale_q);
    proj_kernel<0><<<1024, 256, 0, stream>>>(Xk, Wk, b_K, Kbf, 1.0f);
    proj_kernel<2><<<1024, 256, 0, stream>>>(Xv, Wv, b_V, Vt, 1.0f);

    // attention: 2048 (b,h,qtile) waves / 4 waves per block = 512 blocks
    attn_kernel<<<512, 128, 0, stream>>>(Qbf, Kbf, Vt, lens, Abf);

    // output projection -> f32 d_out
    proj_kernel<3><<<1024, 256, 0, stream>>>(Abf, Wo, b_O, d_out, 1.0f);
}
